// CustomCausalAttention_21603685499692
// MI455X (gfx1250) — matrix operand order repaired and checked by reading, not compiled, not run on hardware
//
#include <hip/hip_runtime.h>

typedef __attribute__((ext_vector_type(16))) _Float16 v16h;
typedef __attribute__((ext_vector_type(8)))  float    v8f;
typedef __attribute__((ext_vector_type(4)))  int      v4i;

#define LDS_AS __attribute__((address_space(3)))
#define GLB_AS __attribute__((address_space(1)))

#if __has_builtin(__builtin_amdgcn_global_load_async_to_lds_b128) && \
    __has_builtin(__builtin_amdgcn_s_wait_asynccnt)
#define USE_ASYNC_LDS 1
#else
#define USE_ASYNC_LDS 0
#endif

namespace {
constexpr int   kB = 2, kH = 16, kN = 2048, kD = 128;
constexpr float kScale = 0.08838834764831845f;   // 1/sqrt(D)
constexpr float kLog2e = 1.4426950408889634f;
constexpr int   BM = 128;   // query rows per workgroup (8 waves x 16)
constexpr int   BN = 32;    // keys staged per iteration
constexpr int   WAVES = 8;
}

union HFrag {
  v16h h;
  uint4 u[2];
};

// Unnormalized RBF-kernel causal attention:
//   P_mn = exp(sm_scale*(2 q.k - |q|^2 - |k|^2)) for n <= m, O = P V.
// S = QK^T and O = PV on v_wmma_f32_16x16x32_f16 (f32 accumulate).
// K/V tiles double-buffered in LDS via async global->LDS DMA (ASYNCcnt).
__global__ __launch_bounds__(256) void rbf_causal_attn_wmma(
    const float* __restrict__ Q, const float* __restrict__ K,
    const float* __restrict__ V, float* __restrict__ O) {
  __shared__ float    sRawK[2][BN][kD];  // raw f32 K tiles (double buffer)
  __shared__ float    sRawV[2][BN][kD];  // raw f32 V tiles (double buffer)
  __shared__ _Float16 sK[BN][kD];        // K tile, row-major f16
  __shared__ _Float16 sVt[kD][BN];       // V tile, transposed f16
  __shared__ float    sKsq[BN];          // |k|^2 per staged key
  __shared__ _Float16 sP[WAVES][16][BN]; // per-wave P staging (C->A repack)

  const int tid  = threadIdx.x;
  const int w    = tid >> 5;
  const int lane = tid & 31;
  const int hi   = lane >> 4;       // which 16-lane half
  const int ln   = lane & 15;

  const int    bh   = blockIdx.y;   // fused (b,h)
  const int    m0   = blockIdx.x * BM + w * 16;
  const size_t base = (size_t)bh * kN * kD;
  const int    jEnd = (blockIdx.x + 1) * BM;   // causal horizon of this block

  // staging assignment: 8 threads per key row, 16 floats each
  const int key = tid >> 3;
  const int d0  = (tid & 7) * 16;
  const float* gk0 = K + base + (size_t)key * kD + d0;
  const float* gv0 = V + base + (size_t)key * kD + d0;

#if USE_ASYNC_LDS
  auto issue_tile = [&](int j0, int buf) {
    const float* gk = gk0 + (size_t)j0 * kD;
    const float* gv = gv0 + (size_t)j0 * kD;
    float* lk = &sRawK[buf][key][d0];
    float* lv = &sRawV[buf][key][d0];
    for (int i = 0; i < 4; ++i) {
      __builtin_amdgcn_global_load_async_to_lds_b128(
          (GLB_AS v4i*)(gk + i * 4), (LDS_AS v4i*)(lk + i * 4), 0, 0);
      __builtin_amdgcn_global_load_async_to_lds_b128(
          (GLB_AS v4i*)(gv + i * 4), (LDS_AS v4i*)(lv + i * 4), 0, 0);
    }
  };
  issue_tile(0, 0);  // prologue: DMA first tile while we set up Q
#endif

  // ---- this wave's 16x128 Q tile -> 4 f16 A-fragments (16x32 each), plus |q|^2
  HFrag aQ[4];
  float qsq = 0.f;
  {
    const float* qrow = Q + base + (size_t)(m0 + ln) * kD;
    for (int c = 0; c < 4; ++c) {
      // A-layout: elems 0..7 <-> K = hi*8 + 0..7 ; elems 8..15 <-> K = 16 + hi*8 + 0..7
      const float4* p0 = reinterpret_cast<const float4*>(qrow + c * 32 + hi * 8);
      const float4* p1 = reinterpret_cast<const float4*>(qrow + c * 32 + 16 + hi * 8);
      float4 x0 = p0[0], x1 = p0[1], x2 = p1[0], x3 = p1[1];
      float xs[16] = {x0.x, x0.y, x0.z, x0.w, x1.x, x1.y, x1.z, x1.w,
                      x2.x, x2.y, x2.z, x2.w, x3.x, x3.y, x3.z, x3.w};
      for (int e = 0; e < 16; ++e) {
        aQ[c].h[e] = (_Float16)xs[e];
        qsq += xs[e] * xs[e];
      }
    }
    qsq += __shfl_xor(qsq, 16);                 // combine the two half-row partials
  }
  float qs[8];                                  // |q|^2 for C/D rows r + hi*8
  for (int r = 0; r < 8; ++r) qs[r] = __shfl(qsq, r + hi * 8);

  v8f acc[8];                                   // O tile: 16 rows x 128 cols
  for (int dc = 0; dc < 8; ++dc) acc[dc] = {};

  for (int j0 = 0; j0 < jEnd; j0 += BN) {
    const int buf = (j0 / BN) & 1;
    (void)buf;
#if USE_ASYNC_LDS
    __builtin_amdgcn_s_wait_asynccnt(0);        // tile j0 raw data landed in LDS
#endif
    __syncthreads();                            // all waves done with prev tile
#if USE_ASYNC_LDS
    if (j0 + BN < jEnd) issue_tile(j0 + BN, buf ^ 1);  // DMA next tile now
#endif
    if (tid < BN) sKsq[tid] = 0.f;
    __syncthreads();

    // ---- convert raw tile -> sK (row-major f16), sVt (transposed f16), |k|^2
    {
      const float* ksrc;
      const float* vsrc;
#if USE_ASYNC_LDS
      ksrc = &sRawK[buf][key][d0];
      vsrc = &sRawV[buf][key][d0];
#else
      ksrc = gk0 + (size_t)j0 * kD;
      vsrc = gv0 + (size_t)j0 * kD;
#endif
      union { _Float16 h[16]; uint4 u[2]; } pk;
      float ks = 0.f;
      const float4* k4 = reinterpret_cast<const float4*>(ksrc);
      for (int i = 0; i < 4; ++i) {
        float4 x = k4[i];
        float xs[4] = {x.x, x.y, x.z, x.w};
        for (int e = 0; e < 4; ++e) {
          pk.h[i * 4 + e] = (_Float16)xs[e];
          ks += xs[e] * xs[e];
        }
      }
      *reinterpret_cast<uint4*>(&sK[key][d0])     = pk.u[0];
      *reinterpret_cast<uint4*>(&sK[key][d0 + 8]) = pk.u[1];
      atomicAdd(&sKsq[key], ks);

      const float4* v4 = reinterpret_cast<const float4*>(vsrc);
      for (int i = 0; i < 4; ++i) {
        float4 x = v4[i];
        sVt[d0 + i * 4 + 0][key] = (_Float16)x.x;
        sVt[d0 + i * 4 + 1][key] = (_Float16)x.y;
        sVt[d0 + i * 4 + 2][key] = (_Float16)x.z;
        sVt[d0 + i * 4 + 3][key] = (_Float16)x.w;
      }
#if !USE_ASYNC_LDS
      if (j0 + BN < jEnd) {                     // fallback: warm caches
        __builtin_prefetch(ksrc + (size_t)BN * kD, 0, 1);
        __builtin_prefetch(vsrc + (size_t)BN * kD, 0, 1);
      }
#endif
    }
    __syncthreads();

    if (j0 > m0 + 15) continue;                 // wave-uniform: tile fully masked

    // ---- S = Q K^T : two 16-key subtiles, 4 WMMAs each over D=128
    for (int t = 0; t < 2; ++t) {
      v8f s = {};
      for (int c = 0; c < 4; ++c) {
        HFrag bK;  // B-layout: elems 0..7 <-> K = hi*8 + 0..7 ; elems 8..15 <-> K = 16 + hi*8 + 0..7
        const uint4* kp4 =
            reinterpret_cast<const uint4*>(&sK[t * 16 + ln][c * 32 + hi * 8]);
        bK.u[0] = kp4[0];
        bK.u[1] = kp4[2];
        s = __builtin_amdgcn_wmma_f32_16x16x32_f16(false, aQ[c].h, false, bK.h,
                                                   (short)0, s, false, false);
      }
      const float ksq   = sKsq[t * 16 + ln];
      const int   nglob = j0 + t * 16 + ln;
      for (int r = 0; r < 8; ++r) {
        const int mglob = m0 + r + hi * 8;
        float x = (2.f * kScale * kLog2e) * s[r] -
                  (kScale * kLog2e) * (qs[r] + ksq);
        float p = (nglob <= mglob) ? __builtin_amdgcn_exp2f(x) : 0.f;
        sP[w][r + hi * 8][t * 16 + ln] = (_Float16)p;  // C-layout -> LDS
      }
    }

    // ---- O += P V (per-wave LDS repack; same-wave DS ops are in order)
    HFrag aP;
    aP.u[0] = *reinterpret_cast<const uint4*>(&sP[w][ln][hi * 8]);
    aP.u[1] = *reinterpret_cast<const uint4*>(&sP[w][ln][16 + hi * 8]);
    for (int dc = 0; dc < 8; ++dc) {
      HFrag bV;
      const uint4* vp4 =
          reinterpret_cast<const uint4*>(&sVt[dc * 16 + ln][hi * 8]);
      bV.u[0] = vp4[0];
      bV.u[1] = vp4[2];
      acc[dc] = __builtin_amdgcn_wmma_f32_16x16x32_f16(false, aP.h, false, bV.h,
                                                       (short)0, acc[dc], false, false);
    }
  }

  // ---- writeback O (16 rows x 128 cols per wave)
  float* orow = O + base;
  for (int r = 0; r < 8; ++r) {
    const size_t m = (size_t)(m0 + r + hi * 8);
    for (int dc = 0; dc < 8; ++dc) {
      orow[m * kD + dc * 16 + ln] = acc[dc][r];
    }
  }
}

extern "C" void kernel_launch(void* const* d_in, const int* in_sizes, int n_in,
                              void* d_out, int out_size, void* d_ws, size_t ws_size,
                              hipStream_t stream) {
  (void)in_sizes; (void)n_in; (void)out_size; (void)d_ws; (void)ws_size;
  const float* q = (const float*)d_in[0];
  const float* k = (const float*)d_in[1];
  const float* v = (const float*)d_in[2];
  float* o = (float*)d_out;
  dim3 grid(kN / BM, kB * kH);
  dim3 block(256);
  hipLaunchKernelGGL(rbf_causal_attn_wmma, grid, block, 0, stream, q, k, v, o);
}
